// YOLOv3Tiny_61254823575660
// MI455X (gfx1250) — matrix operand order repaired and checked by reading, not compiled, not run on hardware
//
#include <hip/hip_runtime.h>

// ---------------------------------------------------------------------------
// YOLO head for MI455X (gfx1250, wave32): all 1x1 convs as f16 WMMA GEMMs
// (V_WMMA_F32_16X16X32_F16, f32 accumulate), f16 activation storage.
// LDS tiles are stored PRE-SWIZZLED in WMMA fragment order so fragment
// loads are contiguous ds_load_b128 pairs (no per-element LDS gathers).
// Each block: 128 threads (4 waves), tile M=16 x N=128, K unrolled by 64
// -> 4 WMMAs per barrier interval, one A fragment per 32-wide k-step.
// ---------------------------------------------------------------------------

typedef __attribute__((ext_vector_type(16))) _Float16 v16h;
typedef __attribute__((ext_vector_type(8)))  float    v8f;
typedef __attribute__((ext_vector_type(8)))  _Float16 h8;
typedef __attribute__((ext_vector_type(4)))  _Float16 h4;

// ---------------- f32 -> f16 convert ----------------
__global__ void cvt_f16_kernel(const float* __restrict__ x,
                               _Float16* __restrict__ y, int n) {
  int i = blockIdx.x * blockDim.x + threadIdx.x;
  if (i < n) y[i] = (_Float16)x[i];
}

// ---------------- fused 1x1 conv (GEMM) via WMMA ----------------
// Y[b,o,hw] = act( scale[o] * (sum_c W[o,c] * X[b,c,hw]) + bias[o] )
// A-fragment element e of lane group g holds K = base(v)+p+8g (CDNA5 ISA);
// inverse: g = (k>>3)&1 ; e = (k&7) | ((k&16)>>1)   (aligned-8 k-runs -> e-runs)
// B-fragment element e of lane group g holds K = (e&7) + 8g + 16*(e>>3).
__global__ __launch_bounds__(128) void conv1x1_wmma_kernel(
    const _Float16* __restrict__ X, const float* __restrict__ W,
    const float* __restrict__ scale, const float* __restrict__ bias,
    _Float16* __restrict__ Yh, float* __restrict__ Yf,
    int Bn, int Cin, int Cout, int HW, int leaky)
{
  // fragment-order tiles for two 32-wide k-steps
  __shared__ __align__(16) _Float16 a_sw[2][32][24];   // [ks][g*16+m][e]  (pad 24)
  __shared__ __align__(16) _Float16 b_sw[2][128][48];  // [ks][n][k 0..31] (pad 48)

  const int tid  = threadIdx.x;
  const int lane = tid & 31;
  const int wv   = tid >> 5;                  // wave id -> 32-wide N slice
  const int oBase = blockIdx.y * 16;
  const int nBase = blockIdx.x * 128;
  const int Ntot  = Bn * HW;

  const int m  = lane & 15;
  const int g  = lane >> 4;

  // fast path: whole 128-wide n-range inside one batch image, 8B aligned
  const int b0  = nBase / HW;
  const int hw0 = nBase - b0 * HW;
  const bool fast = (nBase + 128 <= Ntot) &&
                    ((nBase + 127) / HW == b0) &&
                    ((hw0 & 3) == 0);

  v8f acc[2] = {{}, {}};

  for (int k0 = 0; k0 < Cin; k0 += 64) {
    // ---- stage weights: 2 x (16x32) f32 -> f16, written in fragment order ----
    {
      int idx = tid * 8;                  // 1024 halves total
      int ks  = idx >> 9;
      int rem = idx & 511;
      int mm  = rem >> 5;
      int kk  = rem & 31;                 // aligned-8 run: same g, contiguous e
      int gg  = (kk >> 3) & 1;
      int e0  = (kk & 7) | ((kk & 16) >> 1);
      float4 wa = {0.f, 0.f, 0.f, 0.f}, wb = {0.f, 0.f, 0.f, 0.f};
      if (oBase + mm < Cout) {
        const float* wp = W + (size_t)(oBase + mm) * Cin + k0 + ks * 32 + kk;
        wa = *(const float4*)(wp);
        wb = *(const float4*)(wp + 4);
      }
      h4 lo = {(_Float16)wa.x, (_Float16)wa.y, (_Float16)wa.z, (_Float16)wa.w};
      h4 hi = {(_Float16)wb.x, (_Float16)wb.y, (_Float16)wb.z, (_Float16)wb.w};
      *(h4*)&a_sw[ks][gg * 16 + mm][e0]     = lo;
      *(h4*)&a_sw[ks][gg * 16 + mm][e0 + 4] = hi;
    }
    // ---- stage activations transposed: b_sw[ks][n][k], 2 x (32k x 128n) ----
    if (fast) {
      const _Float16* base = X + (size_t)(b0 * Cin + k0) * HW + hw0;
      for (int it = 0; it < 16; ++it) {
        int slot = tid + it * 128;        // 2048 h4 slots
        int ks   = slot >> 10;
        int rem  = slot & 1023;
        int kr   = rem >> 5;
        int nseg = rem & 31;
        h4 v = *(const h4*)(base + (size_t)(ks * 32 + kr) * HW + nseg * 4);
        b_sw[ks][nseg * 4 + 0][kr] = v[0];
        b_sw[ks][nseg * 4 + 1][kr] = v[1];
        b_sw[ks][nseg * 4 + 2][kr] = v[2];
        b_sw[ks][nseg * 4 + 3][kr] = v[3];
      }
    } else {
      for (int idx = tid; idx < 8192; idx += 128) {
        int ks  = idx >> 12;
        int rem = idx & 4095;
        int kr  = rem >> 7;
        int j   = rem & 127;
        int n   = nBase + j;
        _Float16 v = (_Float16)0.f;
        if (n < Ntot) {
          int b  = n / HW;
          int hw = n - b * HW;
          v = X[(size_t)(b * Cin + k0 + ks * 32 + kr) * HW + hw];
        }
        b_sw[ks][j][kr] = v;
      }
    }
    __syncthreads();

    // ---- 2 k-steps x 2 n-subtiles = 4 WMMAs, one A fragment per k-step ----
#pragma unroll
    for (int ks = 0; ks < 2; ++ks) {
      h8 alo = *(const h8*)&a_sw[ks][g * 16 + m][0];
      h8 ahi = *(const h8*)&a_sw[ks][g * 16 + m][8];
      v16h af = __builtin_shufflevector(alo, ahi, 0, 1, 2, 3, 4, 5, 6, 7,
                                        8, 9, 10, 11, 12, 13, 14, 15);
#pragma unroll
      for (int nsub = 0; nsub < 2; ++nsub) {
        const _Float16* bp = &b_sw[ks][wv * 32 + nsub * 16 + m][8 * g];
        h8 blo = *(const h8*)(bp);
        h8 bhi = *(const h8*)(bp + 16);
        v16h bf = __builtin_shufflevector(blo, bhi, 0, 1, 2, 3, 4, 5, 6, 7,
                                          8, 9, 10, 11, 12, 13, 14, 15);
        acc[nsub] = __builtin_amdgcn_wmma_f32_16x16x32_f16(
            false, af, false, bf, (short)0, acc[nsub], false, false);
      }
    }
    __syncthreads();
  }

  // ---- epilogue: folded-BN + LeakyReLU (or plain bias), f16 or f32 out ----
#pragma unroll
  for (int nsub = 0; nsub < 2; ++nsub) {
    int n = nBase + wv * 32 + nsub * 16 + m;
    if (n < Ntot) {
      int b  = n / HW;
      int hw = n - b * HW;
#pragma unroll
      for (int v = 0; v < 8; ++v) {
        int o = oBase + v + 8 * g;        // C/D layout: M = vgpr + 8*laneGroup
        if (o < Cout) {
          float val = acc[nsub][v];
          val = (scale ? scale[o] : 1.0f) * val + bias[o];
          if (leaky) val = (val > 0.f) ? val : 0.1f * val;
          size_t off = ((size_t)b * Cout + o) * HW + hw;
          if (Yh) Yh[off] = (_Float16)val;
          else    Yf[off] = val;
        }
      }
    }
  }
}

// ---------------- SPP: write [x, mp5, mp9, mp13] concat directly ----------------
__global__ void spp_concat_kernel(const _Float16* __restrict__ x,
                                  _Float16* __restrict__ y,
                                  int Bn, int C, int H, int W) {
  int i = blockIdx.x * blockDim.x + threadIdx.x;
  int total = Bn * C * H * W;
  if (i >= total) return;
  int w = i % W;
  int h = (i / W) % H;
  int c = (i / (W * H)) % C;
  int b = i / (W * H * C);
  const _Float16* p = x + (size_t)(b * C + c) * H * W;
  float v0 = (float)p[h * W + w];
  float m5 = -1e30f, m9 = -1e30f, m13 = -1e30f;
  for (int dy = -6; dy <= 6; ++dy) {
    int hh = h + dy;
    if (hh < 0 || hh >= H) continue;
    for (int dx = -6; dx <= 6; ++dx) {
      int ww = w + dx;
      if (ww < 0 || ww >= W) continue;
      float v = (float)p[hh * W + ww];
      if (dy >= -2 && dy <= 2 && dx >= -2 && dx <= 2) m5 = fmaxf(m5, v);
      if (dy >= -4 && dy <= 4 && dx >= -4 && dx <= 4) m9 = fmaxf(m9, v);
      m13 = fmaxf(m13, v);
    }
  }
  size_t HWp  = (size_t)H * W;
  size_t base = (size_t)b * 4 * C * HWp + (size_t)h * W + w;
  y[base + (size_t)(0 * C + c) * HWp] = (_Float16)v0;
  y[base + (size_t)(1 * C + c) * HWp] = (_Float16)m5;
  y[base + (size_t)(2 * C + c) * HWp] = (_Float16)m9;
  y[base + (size_t)(3 * C + c) * HWp] = (_Float16)m13;
}

// ---------------- concat(up2(a), b2) ----------------
__global__ void up2_concat_kernel(const _Float16* __restrict__ a,
                                  const _Float16* __restrict__ b2,
                                  _Float16* __restrict__ y,
                                  int Bn, int C1, int C2, int H, int W) {
  int i = blockIdx.x * blockDim.x + threadIdx.x;
  int C = C1 + C2;
  int total = Bn * C * H * W;
  if (i >= total) return;
  int w = i % W;
  int h = (i / W) % H;
  int c = (i / (W * H)) % C;
  int b = i / (W * H * C);
  _Float16 v;
  if (c < C1)
    v = a[(size_t)(b * C1 + c) * (H / 2) * (W / 2) + (h >> 1) * (W / 2) + (w >> 1)];
  else
    v = b2[(size_t)(b * C2 + (c - C1)) * H * W + h * W + w];
  y[i] = v;
}

// ---------------- concat(maxpool3x3s2p1(a), b2) ----------------
__global__ void mp2_concat_kernel(const _Float16* __restrict__ a,
                                  const _Float16* __restrict__ b2,
                                  _Float16* __restrict__ y,
                                  int Bn, int C1, int C2,
                                  int Hin, int Win, int H, int W) {
  int i = blockIdx.x * blockDim.x + threadIdx.x;
  int C = C1 + C2;
  int total = Bn * C * H * W;
  if (i >= total) return;
  int w = i % W;
  int h = (i / W) % H;
  int c = (i / (W * H)) % C;
  int b = i / (W * H * C);
  _Float16 v;
  if (c < C1) {
    const _Float16* p = a + (size_t)(b * C1 + c) * Hin * Win;
    float m = -1e30f;
    int h0 = h * 2 - 1, w0 = w * 2 - 1;
#pragma unroll
    for (int dy = 0; dy < 3; ++dy) {
      int hh = h0 + dy;
      if (hh < 0 || hh >= Hin) continue;
#pragma unroll
      for (int dx = 0; dx < 3; ++dx) {
        int ww = w0 + dx;
        if (ww < 0 || ww >= Win) continue;
        m = fmaxf(m, (float)p[hh * Win + ww]);
      }
    }
    v = (_Float16)m;
  } else {
    v = b2[(size_t)(b * C2 + (c - C1)) * H * W + h * W + w];
  }
  y[i] = v;
}

// ---------------- decode: boxes + softmax(cls)*sigmoid(conf) ----------------
__global__ void decode_kernel(const float* __restrict__ d1,
                              const float* __restrict__ d2,
                              const float* __restrict__ d3,
                              float* __restrict__ boxes,
                              float* __restrict__ scores, int Bn) {
  const int N0 = 76 * 76 * 3, N1 = 38 * 38 * 3, N2 = 19 * 19 * 3;
  const int NT = N0 + N1 + N2;  // 22743
  int idx = blockIdx.x * blockDim.x + threadIdx.x;
  if (idx >= Bn * NT) return;
  int b = idx / NT;
  int n = idx - b * NT;

  const float* P; int HWl, Wl, lvl, local; float stridef;
  if (n < N0)           { P = d1; local = n;           HWl = 5776; Wl = 76; stridef =  8.f; lvl = 0; }
  else if (n < N0 + N1) { P = d2; local = n - N0;      HWl = 1444; Wl = 38; stridef = 16.f; lvl = 1; }
  else                  { P = d3; local = n - N0 - N1; HWl =  361; Wl = 19; stridef = 32.f; lvl = 2; }
  int hw = local / 3;
  int a  = local - 3 * hw;
  const float* Q = P + (size_t)b * 75 * HWl + hw;   // det layout [b,75,HWl]

  float conf = 1.f / (1.f + __expf(-Q[(size_t)a * HWl]));

  float e[20];
  float mx = -1e30f;
#pragma unroll
  for (int c = 0; c < 20; ++c) {
    e[c] = Q[(size_t)(3 + a * 20 + c) * HWl];
    mx = fmaxf(mx, e[c]);
  }
  float s = 0.f;
#pragma unroll
  for (int c = 0; c < 20; ++c) { e[c] = __expf(e[c] - mx); s += e[c]; }
  float kmul = conf / s;

  float t0 = Q[(size_t)(63 + a * 4 + 0) * HWl];
  float t1 = Q[(size_t)(63 + a * 4 + 1) * HWl];
  float t2 = Q[(size_t)(63 + a * 4 + 2) * HWl];
  float t3 = Q[(size_t)(63 + a * 4 + 3) * HWl];

  const float ANCH[3][3][2] = {{{10,13},{16,30},{33,23}},
                               {{30,61},{62,45},{59,119}},
                               {{116,90},{156,198},{373,326}}};
  int gx = hw % Wl, gy = hw / Wl;
  float cx = (1.f / (1.f + __expf(-t0)) + (float)gx) * stridef;
  float cy = (1.f / (1.f + __expf(-t1)) + (float)gy) * stridef;
  float bw = __expf(t2) * ANCH[lvl][a][0];
  float bh = __expf(t3) * ANCH[lvl][a][1];
  const float inv = 1.f / 608.f;
  float x1 = fminf(fmaxf((cx - 0.5f * bw) * inv, 0.f), 1.f);
  float y1 = fminf(fmaxf((cy - 0.5f * bh) * inv, 0.f), 1.f);
  float x2 = fminf(fmaxf((cx + 0.5f * bw) * inv, 0.f), 1.f);
  float y2 = fminf(fmaxf((cy + 0.5f * bh) * inv, 0.f), 1.f);

  size_t bo = (size_t)b * NT + n;
  boxes[bo * 4 + 0] = x1;
  boxes[bo * 4 + 1] = y1;
  boxes[bo * 4 + 2] = x2;
  boxes[bo * 4 + 3] = y2;
#pragma unroll
  for (int c = 0; c < 20; ++c) scores[bo * 20 + c] = e[c] * kmul;
}

// ---------------------------------------------------------------------------
extern "C" void kernel_launch(void* const* d_in, const int* in_sizes, int n_in,
                              void* d_out, int out_size, void* d_ws, size_t ws_size,
                              hipStream_t stream) {
  (void)in_sizes; (void)n_in; (void)out_size; (void)ws_size;
  const int Bn = 8;
  const int HW3 = 76 * 76, HW4 = 38 * 38, HW5 = 19 * 19;

  // d_in order (setup_inputs dict order, params flattened in insertion order):
  // 0:c3 1:c4 2:c5
  // 3..5:   spp_pre{w,s,b}      6..8:   spp_post{w,s,b}
  // 9..11:  head_conv_0{w,s,b}  12..14: head_csp_0{w,s,b}
  // 15..17: head_conv_1{w,s,b}  18..20: head_csp_1{w,s,b}
  // 21..23: head_csp_2{w,s,b}   24..26: head_csp_3{w,s,b}
  // 27,28: det_1{w,b}  29,30: det_2{w,b}  31,32: det_3{w,b}

  // ---- workspace layout (bytes), lifetime-based reuse; peak = 71,529,984 B ----
  char* ws = (char*)d_ws;
  constexpr size_t OFF_C3H  = 0;          // f16  8x128x5776
  constexpr size_t OFF_C4H  = 11829248;   // f16  8x256x1444
  constexpr size_t OFF_C5H  = 17743872;   // f16  8x512x361
  constexpr size_t OFF_TPRE = 20701184;   // f16  8x256x361
  constexpr size_t OFF_TSPP = 22179840;   // f16  8x1024x361
  constexpr size_t OFF_C5S  = 28094464;   // f16  8x512x361
  constexpr size_t OFF_C6   = 20701184;   // f16  8x256x361   (reuse TPRE)
  constexpr size_t OFF_CAT0 = 31051776;   // f16  8x512x1444
  constexpr size_t OFF_C9   = 22179840;   // f16  8x256x1444  (reuse TSPP)
  constexpr size_t OFF_C10  = 42881024;   // f16  8x128x1444
  constexpr size_t OFF_CAT1 = 45838336;   // f16  8x256x5776
  constexpr size_t OFF_C13  = 31051776;   // f16  8x128x5776  (reuse CAT0)
  constexpr size_t OFF_CAT2 = 22179840;   // f16  8x256x1444  (reuse C9)
  constexpr size_t OFF_C16  = 45838336;   // f16  8x256x1444  (reuse CAT1)
  constexpr size_t OFF_CAT3 = 51752960;   // f16  8x512x361
  constexpr size_t OFF_C19  = 54710272;   // f16  8x512x361
  constexpr size_t OFF_DET1 = 57667584;   // f32  8x75x5776
  constexpr size_t OFF_DET2 = 0;          // f32  8x75x1444   (reuse C3H)
  constexpr size_t OFF_DET3 = 3465600;    // f32  8x75x361

  _Float16* c3h  = (_Float16*)(ws + OFF_C3H);
  _Float16* c4h  = (_Float16*)(ws + OFF_C4H);
  _Float16* c5h  = (_Float16*)(ws + OFF_C5H);
  _Float16* tpre = (_Float16*)(ws + OFF_TPRE);
  _Float16* tspp = (_Float16*)(ws + OFF_TSPP);
  _Float16* c5s  = (_Float16*)(ws + OFF_C5S);
  _Float16* c6   = (_Float16*)(ws + OFF_C6);
  _Float16* cat0 = (_Float16*)(ws + OFF_CAT0);
  _Float16* c9   = (_Float16*)(ws + OFF_C9);
  _Float16* c10  = (_Float16*)(ws + OFF_C10);
  _Float16* cat1 = (_Float16*)(ws + OFF_CAT1);
  _Float16* c13  = (_Float16*)(ws + OFF_C13);
  _Float16* cat2 = (_Float16*)(ws + OFF_CAT2);
  _Float16* c16  = (_Float16*)(ws + OFF_C16);
  _Float16* cat3 = (_Float16*)(ws + OFF_CAT3);
  _Float16* c19  = (_Float16*)(ws + OFF_C19);
  float*    det1 = (float*)(ws + OFF_DET1);
  float*    det2 = (float*)(ws + OFF_DET2);
  float*    det3 = (float*)(ws + OFF_DET3);

  auto cvt = [&](const void* src, _Float16* dst, int n) {
    cvt_f16_kernel<<<(n + 255) / 256, 256, 0, stream>>>((const float*)src, dst, n);
  };
  auto conv = [&](const _Float16* Xp, int wi, int si, int bi,
                  _Float16* Yh, float* Yf, int Cin, int Cout, int HW, int leaky) {
    dim3 grid((unsigned)((Bn * HW + 127) / 128), (unsigned)((Cout + 15) / 16));
    conv1x1_wmma_kernel<<<grid, 128, 0, stream>>>(
        Xp, (const float*)d_in[wi],
        si >= 0 ? (const float*)d_in[si] : nullptr,
        (const float*)d_in[bi], Yh, Yf, Bn, Cin, Cout, HW, leaky);
  };
  auto ew = [&](int total) { return dim3((unsigned)((total + 255) / 256)); };

  // ---- input f32 -> f16 ----
  cvt(d_in[0], c3h, Bn * 128 * HW3);
  cvt(d_in[1], c4h, Bn * 256 * HW4);
  cvt(d_in[2], c5h, Bn * 512 * HW5);

  // ---- SPP stage ----
  conv(c5h, 3, 4, 5, tpre, nullptr, 512, 256, HW5, 1);                 // spp_pre
  spp_concat_kernel<<<ew(Bn * 256 * HW5), 256, 0, stream>>>(tpre, tspp, Bn, 256, 19, 19);
  conv(tspp, 6, 7, 8, c5s, nullptr, 1024, 512, HW5, 1);                // spp_post
  conv(c5s, 9, 10, 11, c6, nullptr, 512, 256, HW5, 1);                 // head_conv_0

  // ---- FPN up path ----
  up2_concat_kernel<<<ew(Bn * 512 * HW4), 256, 0, stream>>>(c6, c4h, cat0, Bn, 256, 256, 38, 38);
  conv(cat0, 12, 13, 14, c9, nullptr, 512, 256, HW4, 1);               // head_csp_0
  conv(c9, 15, 16, 17, c10, nullptr, 256, 128, HW4, 1);                // head_conv_1
  up2_concat_kernel<<<ew(Bn * 256 * HW3), 256, 0, stream>>>(c10, c3h, cat1, Bn, 128, 128, 76, 76);
  conv(cat1, 18, 19, 20, c13, nullptr, 256, 128, HW3, 1);              // head_csp_1

  // ---- PAN down path ----
  mp2_concat_kernel<<<ew(Bn * 256 * HW4), 256, 0, stream>>>(c13, c10, cat2, Bn, 128, 128, 76, 76, 38, 38);
  conv(cat2, 21, 22, 23, c16, nullptr, 256, 256, HW4, 1);              // head_csp_2
  mp2_concat_kernel<<<ew(Bn * 512 * HW5), 256, 0, stream>>>(c16, c6, cat3, Bn, 256, 256, 38, 38, 19, 19);
  conv(cat3, 24, 25, 26, c19, nullptr, 512, 512, HW5, 1);              // head_csp_3

  // ---- detection heads (f32 out, no activation/scale) ----
  conv(c13, 27, -1, 28, nullptr, det1, 128, 75, HW3, 0);
  conv(c16, 29, -1, 30, nullptr, det2, 256, 75, HW4, 0);
  conv(c19, 31, -1, 32, nullptr, det3, 512, 75, HW5, 0);

  // ---- decode: boxes [8,22743,4] then scores [8,22743,20] into d_out ----
  const int NT = 22743;
  float* boxes  = (float*)d_out;
  float* scores = (float*)d_out + (size_t)Bn * NT * 4;
  decode_kernel<<<(Bn * NT + 255) / 256, 256, 0, stream>>>(det1, det2, det3, boxes, scores, Bn);
}
